// ScaledDotProductAttention_52166672777934
// MI455X (gfx1250) — matrix operand order repaired and checked by reading, not compiled, not run on hardware
//
#include <hip/hip_runtime.h>
#include <math.h>

typedef _Float16 v16h __attribute__((ext_vector_type(16)));
typedef _Float16 v8h  __attribute__((ext_vector_type(8)));
typedef _Float16 v2h  __attribute__((ext_vector_type(2)));
typedef float    v8f  __attribute__((ext_vector_type(8)));

#define ATT_B  16
#define ATT_SQ 2048
#define ATT_SK 2048
#define ATT_D  64
#define QTILE  16

// LDS layout (float units):
//   S[16*2048] | rsum[16] | cpart[8*16*16] | VT (2 buffers, f16 64x72 each)
#define SMEM_S_ELEMS   (QTILE * ATT_SK)            // 32768
#define OFF_RSUM       SMEM_S_ELEMS                // 16 floats
#define OFF_CPART      (OFF_RSUM + 16)             // 2048 floats
#define OFF_VT         (OFF_CPART + 8 * 16 * 16)   // f16 region starts here
#define VT_STRIDE      72                          // f16 per row (144B, 16B-aligned)
#define VT_BUF_H       (64 * VT_STRIDE)            // f16 per buffer = 4608
#define SMEM_BYTES     ((OFF_VT * 4) + 2 * VT_BUF_H * 2)

__global__ __launch_bounds__(256)
void sdpa_wmma_kernel(const float* __restrict__ Q,
                      const float* __restrict__ K,
                      const float* __restrict__ V,
                      const unsigned char* __restrict__ mask,
                      float* __restrict__ ctx_out,
                      float* __restrict__ attn_out) {
    extern __shared__ float smem[];
    float*     S      = smem;                        // [16][2048] scores -> exp -> P
    float*     rsum   = smem + OFF_RSUM;             // [16] 1/sum
    float*     cpart  = smem + OFF_CPART;            // [8][16][16]
    _Float16*  vtbase = (_Float16*)(smem + OFF_VT);  // 2 x [64][72] f16 (V^T chunks)

    const int bz  = blockIdx.y;
    const int q0  = blockIdx.x * QTILE;
    const int tid = threadIdx.x;
    const int wv  = tid >> 5;                        // wave 0..7
    const int ln  = tid & 15;                        // lane within half
    const int h   = (tid >> 4) & 1;                  // lane half

    const float scale   = 0.125f;                    // 1/sqrt(64)
    const float NEG_INF = -__builtin_inff();

    // ---- Phase 0: Q A-fragments (16-bit A 16x32 layout; two linear runs) --
    v16h aq0, aq1;
    {
        const float* qrow = Q + ((size_t)(bz * ATT_SQ) + q0 + ln) * ATT_D;
        #pragma unroll
        for (int e = 0; e < 8; ++e) {
            aq0[e]     = (_Float16)qrow[8 * h + e];
            aq0[e + 8] = (_Float16)qrow[16 + 8 * h + e];
            aq1[e]     = (_Float16)qrow[32 + 8 * h + e];
            aq1[e + 8] = (_Float16)qrow[48 + 8 * h + e];
        }
    }

    // ---- Phase 1: S = mask(scale * Q K^T) -> LDS --------------------------
    #pragma unroll 1
    for (int t = 0; t < 16; ++t) {
        const int kt   = t * 8 + wv;                 // column tile 0..127
        const int krow = kt * 16 + ln;
        const float* krp = K + ((size_t)(bz * ATT_SK) + krow) * ATT_D;

        // branchless prefetch of next K tile row (wraps in-bounds at t==15)
        {
            const int ktn = ((t + 1) & 15) * 8 + wv;
            __builtin_prefetch(K + ((size_t)(bz * ATT_SK) + ktn * 16 + ln) * ATT_D, 0, 1);
        }

        // B fragment in table order: bk0[e] = K[krow][8h+e], bk0[e+8] = K[krow][16+8h+e], bk1 at d+32
        v16h bk0, bk1;
        #pragma unroll
        for (int e = 0; e < 8; ++e) {
            bk0[e]     = (_Float16)krp[8 * h + e];
            bk0[e + 8] = (_Float16)krp[16 + 8 * h + e];
            bk1[e]     = (_Float16)krp[32 + 8 * h + e];
            bk1[e + 8] = (_Float16)krp[48 + 8 * h + e];
        }

        v8f c = {};
        c = __builtin_amdgcn_wmma_f32_16x16x32_f16(false, aq0, false, bk0,
                                                   (short)0, c, false, false);
        c = __builtin_amdgcn_wmma_f32_16x16x32_f16(false, aq1, false, bk1,
                                                   (short)0, c, false, false);

        const int kg = kt * 16 + ln;
        #pragma unroll
        for (int r = 0; r < 8; ++r) {
            int m = r + 8 * h;                       // C/D: VGPR r -> M = r + 8*half
            float s = c[r] * scale;
            if (mask[((size_t)(bz * ATT_SQ) + q0 + m) * ATT_SK + kg]) s = NEG_INF;
            S[m * ATT_SK + kg] = s;
        }
    }
    __syncthreads();

    // ---- Phase 2a: row max (16 lanes/row, float4 sweeps, wave shuffles) ---
    const int row = tid >> 4;
    const int sub = tid & 15;
    float4* srow4 = (float4*)(S + row * ATT_SK);
    float mx = NEG_INF;
    #pragma unroll 4
    for (int i = 0; i < 32; ++i) {
        float4 v = srow4[sub + 16 * i];
        mx = fmaxf(mx, fmaxf(fmaxf(v.x, v.y), fmaxf(v.z, v.w)));
    }
    #pragma unroll
    for (int off = 8; off >= 1; off >>= 1)
        mx = fmaxf(mx, __shfl_xor(mx, off, 32));

    // ---- Phase 2b: exp in place + row sum ---------------------------------
    float sum = 0.0f;
    #pragma unroll 4
    for (int i = 0; i < 32; ++i) {
        float4 v = srow4[sub + 16 * i];
        v.x = __expf(v.x - mx); v.y = __expf(v.y - mx);
        v.z = __expf(v.z - mx); v.w = __expf(v.w - mx);
        sum += (v.x + v.y) + (v.z + v.w);
        srow4[sub + 16 * i] = v;
    }
    #pragma unroll
    for (int off = 8; off >= 1; off >>= 1)
        sum += __shfl_xor(sum, off, 32);
    if (sub == 0) rsum[row] = 1.0f / sum;
    __syncthreads();

    // ---- Phase 2c: normalize + coalesced b128 attn stream-out -------------
    {
        float4* s4 = (float4*)S;
        float4* a4 = (float4*)(attn_out + ((size_t)(bz * ATT_SQ) + q0) * ATT_SK);
        #pragma unroll 4
        for (int i = 0; i < 32; ++i) {
            int id4 = tid + 256 * i;                 // 0..8191, row = id4>>9
            float r = rsum[id4 >> 9];
            float4 v = s4[id4];
            v.x *= r; v.y *= r; v.z *= r; v.w *= r;
            s4[id4] = v;
            a4[id4] = v;                             // attn tile is row-flat == S
        }
    }
    __syncthreads();

    // ---- Phase 3: context = P @ V, V chunk staged transposed as f16 -------
    const int nt = wv & 3;                           // output d-tile
    const int kw = (wv >> 2) * 32;                   // this wave's k-offset in chunk
    v8f c = {};
    #pragma unroll 1
    for (int it = 0; it < 32; ++it) {
        const int kbase = it * 64;
        _Float16* vt = vtbase + (it & 1) * VT_BUF_H;

        // stage 64(k) x 64(d) chunk -> VT[d][k] f16, coalesced b64 loads
        #pragma unroll
        for (int r2 = 0; r2 < 4; ++r2) {
            int id = tid + 256 * r2;                 // 0..1023
            int kp = id >> 5, dp = id & 31;
            const float* vp = V + ((size_t)(bz * ATT_SK) + kbase + 2 * kp) * ATT_D + 2 * dp;
            float a0 = vp[0], a1 = vp[1];
            float b0 = vp[ATT_D], b1 = vp[ATT_D + 1];
            v2h p0 = { (_Float16)a0, (_Float16)b0 };
            v2h p1 = { (_Float16)a1, (_Float16)b1 };
            *(v2h*)(vt + (2 * dp)     * VT_STRIDE + 2 * kp) = p0;
            *(v2h*)(vt + (2 * dp + 1) * VT_STRIDE + 2 * kp) = p1;
        }
        __syncthreads();

        // A fragment: two contiguous float4-pairs of P per lane
        const float* prow = S + ln * ATT_SK + kbase + kw;
        float4 pA = *(const float4*)(prow + 8 * h);
        float4 pB = *(const float4*)(prow + 8 * h + 4);
        float4 pC = *(const float4*)(prow + 16 + 8 * h);
        float4 pD = *(const float4*)(prow + 16 + 8 * h + 4);
        v16h ap;
        #pragma unroll
        for (int i = 0; i < 4; ++i) {
            ap[i]      = (_Float16)((const float*)&pA)[i];
            ap[i + 4]  = (_Float16)((const float*)&pB)[i];
            ap[i + 8]  = (_Float16)((const float*)&pC)[i];
            ap[i + 12] = (_Float16)((const float*)&pD)[i];
        }

        // B fragment: two 8-f16 runs from VT row at k=8h and k=16+8h (two b128 LDS loads)
        const _Float16* vrow = vt + (nt * 16 + ln) * VT_STRIDE + kw + 8 * h;
        v8h blo = *(const v8h*)(vrow);
        v8h bhi = *(const v8h*)(vrow + 16);
        v16h bv = __builtin_shufflevector(blo, bhi,
                    0, 1, 2, 3, 4, 5, 6, 7, 8, 9, 10, 11, 12, 13, 14, 15);

        c = __builtin_amdgcn_wmma_f32_16x16x32_f16(false, ap, false, bv,
                                                   (short)0, c, false, false);
    }
    {
        float* my = cpart + wv * 256;
        #pragma unroll
        for (int r = 0; r < 8; ++r)
            my[(r + 8 * h) * 16 + ln] = c[r];
    }
    __syncthreads();

    // ---- Phase 4: reduce k-halves, coalesced context write ----------------
    {
        float* cout = ctx_out + ((size_t)(bz * ATT_SQ) + q0) * ATT_D;
        #pragma unroll
        for (int i = 0; i < 4; ++i) {
            int idx = tid + 256 * i;                 // 0..1023
            int m = idx >> 6, d = idx & 63;
            int t2 = d >> 4, n = d & 15;
            cout[(size_t)m * ATT_D + d] =
                cpart[t2 * 256 + m * 16 + n] + cpart[(t2 + 4) * 256 + m * 16 + n];
        }
    }
}

extern "C" void kernel_launch(void* const* d_in, const int* in_sizes, int n_in,
                              void* d_out, int out_size, void* d_ws, size_t ws_size,
                              hipStream_t stream) {
    (void)in_sizes; (void)n_in; (void)d_ws; (void)ws_size; (void)out_size;
    const float* Q = (const float*)d_in[0];
    const float* K = (const float*)d_in[1];
    const float* V = (const float*)d_in[2];
    const unsigned char* mask = (const unsigned char*)d_in[3];

    float* ctx  = (float*)d_out;
    float* attn = (float*)d_out + (size_t)ATT_B * ATT_SQ * ATT_D;

    dim3 grid(ATT_SQ / QTILE, ATT_B);
    dim3 block(256);
    sdpa_wmma_kernel<<<grid, block, (size_t)SMEM_BYTES, stream>>>(Q, K, V, mask, ctx, attn);
}
